// GNNSynEncoder_9251359555634
// MI455X (gfx1250) — matrix operand order repaired and checked by reading, not compiled, not run on hardware
//
#include <hip/hip_runtime.h>

// ---------- types for WMMA ----------
typedef __attribute__((ext_vector_type(16))) __bf16        v16bf;
typedef __attribute__((ext_vector_type(8)))  float         v8f;
typedef __attribute__((ext_vector_type(4)))  unsigned int  v4u;

union Frag16 {            // 8 VGPRs of packed bf16 pairs
    v16bf v;
    unsigned int u[8];
    v4u q[2];
};

#define N_NODES 50000
#define N_EDGES 800000
#define IN_DIM 128
#define EMB 128
#define HID 256           // 2*EMB
#define BN_EPS 1e-5f

static __device__ __forceinline__ unsigned short f2bf(float f) {
    unsigned int u = __float_as_uint(f);
    unsigned int r = (u + 0x7FFFu + ((u >> 16) & 1u)) >> 16;   // RNE
    return (unsigned short)r;
}

// Load the 4 B fragments of one K-step in a single volatile asm block:
// 8x ds_load_b128 issued back-to-back, ONE s_wait_dscnt. The K-step
// displacement is a compile-time offset: immediate, so no per-iteration
// address VALU. Early-clobber outputs cannot alias the address inputs,
// and volatility pins the loads inside the M-tile loop (no LICM/spills).
static __device__ __forceinline__ void ldsB_load4(Frag16* bf,
                                                  unsigned a0, unsigned a1,
                                                  unsigned a2, unsigned a3,
                                                  int off) {
    asm volatile(
        "ds_load_b128 %0, %8 offset:%c12\n\t"
        "ds_load_b128 %1, %8 offset:%c13\n\t"
        "ds_load_b128 %2, %9 offset:%c12\n\t"
        "ds_load_b128 %3, %9 offset:%c13\n\t"
        "ds_load_b128 %4, %10 offset:%c12\n\t"
        "ds_load_b128 %5, %10 offset:%c13\n\t"
        "ds_load_b128 %6, %11 offset:%c12\n\t"
        "ds_load_b128 %7, %11 offset:%c13\n\t"
        "s_wait_dscnt 0x0"
        : "=&v"(bf[0].q[0]), "=&v"(bf[0].q[1]),
          "=&v"(bf[1].q[0]), "=&v"(bf[1].q[1]),
          "=&v"(bf[2].q[0]), "=&v"(bf[2].q[1]),
          "=&v"(bf[3].q[0]), "=&v"(bf[3].q[1])
        : "v"(a0), "v"(a1), "v"(a2), "v"(a3), "i"(off), "i"(off + 16));
}

// ---------- utility ----------
__global__ void k_fill_zero(float* __restrict__ p, int n) {
    int i = blockIdx.x * blockDim.x + threadIdx.x;
    if (i < n) p[i] = 0.0f;
}

// W [K x N] fp32 row-major  ->  Wt [N x K] bf16 row-major (pairs contiguous in K)
__global__ void k_transconv(const float* __restrict__ W, unsigned short* __restrict__ Wt,
                            int K, int N) {
    int i = blockIdx.x * blockDim.x + threadIdx.x;
    if (i < K * N) {
        int k = i / N, n = i - k * N;
        Wt[(size_t)n * K + k] = f2bf(W[i]);
    }
}

// h = x * node_adv (broadcast per row)
__global__ void k_init_h(const float* __restrict__ x, const float* __restrict__ node_adv,
                         float* __restrict__ h, int n) {
    int i = blockIdx.x * blockDim.x + threadIdx.x;
    if (i < n) h[i] = x[i] * node_adv[i >> 7];   // IN_DIM = 128
}

// one wave per edge, lane handles float4 (128 feats / 32 lanes)
__global__ void k_scatter(const float* __restrict__ h, const long long* __restrict__ ei,
                          const float* __restrict__ edge_adv, float* __restrict__ agg) {
    int gid  = blockIdx.x * blockDim.x + threadIdx.x;
    int e    = gid >> 5;
    int lane = gid & 31;
    if (e >= N_EDGES) return;
    int src = (int)ei[e];
    int dst = (int)ei[N_EDGES + e];
    float w = edge_adv[e];
    const float4* hr = (const float4*)(h + (size_t)src * EMB);
    float4 m = hr[lane];
    m.x = fmaxf(m.x * w, 0.0f);
    m.y = fmaxf(m.y * w, 0.0f);
    m.z = fmaxf(m.z * w, 0.0f);
    m.w = fmaxf(m.w * w, 0.0f);
    float* o = agg + (size_t)dst * EMB + lane * 4;
    atomicAdd(o + 0, m.x);
    atomicAdd(o + 1, m.y);
    atomicAdd(o + 2, m.z);
    atomicAdd(o + 3, m.w);
}

// z = (1+eps)*h + agg -> bf16 GEMM input
__global__ void k_z_bf16(const float* __restrict__ h, const float* __restrict__ agg,
                         const float* __restrict__ eps_ptr, unsigned short* __restrict__ A,
                         int n) {
    int i = blockIdx.x * blockDim.x + threadIdx.x;
    if (i < n) {
        float s = 1.0f + *eps_ptr;
        A[i] = f2bf(s * h[i] + agg[i]);
    }
}

// ---------------------------------------------------------------------------
// GEMM with LDS-resident weights.
// C[M,N] = A[M,K](bf16) x Bt[N,K](bf16)^T + bias.
// Per M-tile: one clause of global B128 A-loads; per K-step one asm block of
// 8 ds_load_b128 + single dscnt wait feeding 4 back-to-back WMMAs.
// ---------------------------------------------------------------------------
template <int KSTEPS, int NT>
__global__ void k_gemm_ldsB(const unsigned short* __restrict__ A,
                            const unsigned short* __restrict__ Bt,
                            const float* __restrict__ bias,
                            float* __restrict__ C,
                            int M, int N, int nGroups) {
    static_assert(NT == 4, "ldsB_load4 is specialized for NT==4");
    constexpr int K      = KSTEPS * 32;
    constexpr int COLS   = NT * 16;
    constexpr int LDS_DW = COLS * K / 2;          // dwords of packed bf16 pairs
    __shared__ unsigned int ldsB[LDS_DW];

    int grp  = blockIdx.x % nGroups;
    int slot = blockIdx.x / nGroups;
    int blocksPerGroup = gridDim.x / nGroups;
    int wavesPerBlock  = blockDim.x >> 5;
    int waveIn = threadIdx.x >> 5;
    int lane   = threadIdx.x & 31;
    int n0 = grp * COLS;

    // ---- cooperative fill: contiguous [COLS x K] bf16 strip -> LDS ----
    {
        const uint4* src = (const uint4*)(Bt + (size_t)n0 * K);
        uint4* dst = (uint4*)ldsB;
        for (int i = threadIdx.x; i < LDS_DW / 4; i += blockDim.x)
            dst[i] = src[i];
    }
    __syncthreads();

    int laneLo = lane & 15;
    int hi     = lane >> 4;
    // DS instructions take the LDS byte offset = low 32 bits of the generic addr.
    // B-fragment base addresses are loop-invariant; K-step goes in the DS
    // offset immediate (ks*64 bytes).
    unsigned ldsBase = (unsigned)(size_t)(const void*)&ldsB[0];
    unsigned bA0 = ldsBase + 4u * (unsigned)((0 * 16 + laneLo) * (K / 2) + hi * 8);
    unsigned bA1 = ldsBase + 4u * (unsigned)((1 * 16 + laneLo) * (K / 2) + hi * 8);
    unsigned bA2 = ldsBase + 4u * (unsigned)((2 * 16 + laneLo) * (K / 2) + hi * 8);
    unsigned bA3 = ldsBase + 4u * (unsigned)((3 * 16 + laneLo) * (K / 2) + hi * 8);

    int wavesPerGroup = blocksPerGroup * wavesPerBlock;
    int mTiles = M >> 4;

    for (int mTile = slot * wavesPerBlock + waveIn; mTile < mTiles;
         mTile += wavesPerGroup) {
        int m0 = mTile << 4;
        const unsigned int* arow =
            (const unsigned int*)(A + (size_t)(m0 + laneLo) * K);

        // all A fragments for this tile in one global load clause
        Frag16 a[KSTEPS];
#pragma unroll
        for (int ks = 0; ks < KSTEPS; ++ks) {
#pragma unroll
            for (int v = 0; v < 8; ++v) {
                int kb = ks * 32 + 2 * v + hi * 16;
                a[ks].u[v] = arow[kb >> 1];
            }
        }

        v8f acc[NT];
#pragma unroll
        for (int t = 0; t < NT; ++t) acc[t] = (v8f){};

#pragma unroll
        for (int ks = 0; ks < KSTEPS; ++ks) {
            Frag16 bf[4];
            ldsB_load4(bf, bA0, bA1, bA2, bA3, ks * 64);
#pragma unroll
            for (int t = 0; t < NT; ++t) {
                acc[t] = __builtin_amdgcn_wmma_f32_16x16x32_bf16(
                    false, a[ks].v, false, bf[t].v, (short)0, acc[t], false, false);
            }
        }

        // store: VGPR r -> row = m0 + r + 8*hi ; col = n0 + t*16 + (lane&15)
#pragma unroll
        for (int t = 0; t < NT; ++t) {
            int col = n0 + t * 16 + laneLo;
            float bv = bias[col];
#pragma unroll
            for (int r = 0; r < 8; ++r) {
                C[(size_t)(m0 + r + hi * 8) * N + col] = acc[t][r] + bv;
            }
        }
    }
}

// per-column partial sums (block of N threads, coalesced rows)
__global__ void k_colstats(const float* __restrict__ C, float* __restrict__ sum,
                           float* __restrict__ sumsq, int M, int N, int rowsPerBlock) {
    int col = threadIdx.x;
    int r0 = blockIdx.x * rowsPerBlock;
    int r1 = r0 + rowsPerBlock; if (r1 > M) r1 = M;
    float s = 0.0f, s2 = 0.0f;
    for (int r = r0; r < r1; ++r) {
        float v = C[(size_t)r * N + col];
        s += v; s2 += v * v;
    }
    atomicAdd(&sum[col], s);
    atomicAdd(&sumsq[col], s2);
}

__global__ void k_finalize_stats(const float* __restrict__ sum, const float* __restrict__ sumsq,
                                 const float* __restrict__ gamma, const float* __restrict__ beta,
                                 float* __restrict__ scale, float* __restrict__ shift,
                                 int N, float invM) {
    int x = threadIdx.x;
    if (x < N) {
        float m = sum[x] * invM;
        float v = sumsq[x] * invM - m * m;
        float rs = rsqrtf(v + BN_EPS);
        float sc = gamma[x] * rs;
        scale[x] = sc;
        shift[x] = beta[x] - m * sc;
    }
}

// inner BN + relu -> bf16 activations for GEMM2
__global__ void k_bnrelu_bf16(const float* __restrict__ C, const float* __restrict__ scale,
                              const float* __restrict__ shift, unsigned short* __restrict__ A,
                              int n, int N) {
    int i = blockIdx.x * blockDim.x + threadIdx.x;
    if (i < n) {
        int col = i % N;
        float y = fmaf(C[i], scale[col], shift[col]);
        A[i] = f2bf(fmaxf(y, 0.0f));
    }
}

// outer BN (+ optional relu) -> f32 node features (or final output)
__global__ void k_bn_out(const float* __restrict__ C, const float* __restrict__ scale,
                         const float* __restrict__ shift, float* __restrict__ out,
                         int n, int N, int doRelu) {
    int i = blockIdx.x * blockDim.x + threadIdx.x;
    if (i < n) {
        int col = i % N;
        float y = fmaf(C[i], scale[col], shift[col]);
        out[i] = doRelu ? fmaxf(y, 0.0f) : y;
    }
}

extern "C" void kernel_launch(void* const* d_in, const int* in_sizes, int n_in,
                              void* d_out, int out_size, void* d_ws, size_t ws_size,
                              hipStream_t stream) {
    (void)in_sizes; (void)n_in; (void)out_size; (void)ws_size;

    const float*     x        = (const float*)d_in[0];
    const long long* ei       = (const long long*)d_in[1];   // int64 edge_index [2, E]
    const float*     node_adv = (const float*)d_in[2];
    const float*     edge_adv = (const float*)d_in[3];
    // per layer l (9 tensors): W1 b1 g1 be1 W2 b2 eps g_out be_out
    float* out = (float*)d_out;

    // ---- workspace layout ----
    const size_t NH   = (size_t)N_NODES * EMB;     // 6.4M
    const size_t NHID = (size_t)N_NODES * HID;     // 12.8M
    float* h   = (float*)d_ws;                     // [N,128] f32
    float* agg = h + NH;                           // [N,128] f32
    float* C   = agg + NH;                         // [N,256] f32 (reused for [N,128])
    unsigned short* Abf = (unsigned short*)(C + NHID);   // [N,256] bf16
    unsigned short* Wt  = Abf + NHID;              // 6 x 32768 bf16 transposed weights
    float* stats = (float*)(Wt + 6 * 32768);       // sum|sumsq|scale|shift  (4 x 256)
    float* s_sum = stats, * s_sq = stats + 256, * s_scale = stats + 512, * s_shift = stats + 768;

    const int TB = 256;
    const int ROWS_PER_BLK = 500;                  // 100 blocks over 50000 rows
    const int STAT_BLKS = (N_NODES + ROWS_PER_BLK - 1) / ROWS_PER_BLK;

    const int GEMM_BLOCKS = 256;                   // 256 blocks x 256 threads

    // ---- pre-transpose all weights to bf16 [N,K] ----
    for (int l = 0; l < 3; ++l) {
        const float* W1 = (const float*)d_in[4 + l * 9 + 0];   // [128,256]
        const float* W2 = (const float*)d_in[4 + l * 9 + 4];   // [256,128]
        k_transconv<<<(IN_DIM * HID + TB - 1) / TB, TB, 0, stream>>>(W1, Wt + (2 * l) * 32768, IN_DIM, HID);
        k_transconv<<<(HID * EMB + TB - 1) / TB, TB, 0, stream>>>(W2, Wt + (2 * l + 1) * 32768, HID, EMB);
    }

    // ---- h = x * node_adv ----
    k_init_h<<<(int)((NH + TB - 1) / TB), TB, 0, stream>>>(x, node_adv, h, (int)NH);

    for (int l = 0; l < 3; ++l) {
        const float* b1     = (const float*)d_in[4 + l * 9 + 1];
        const float* g1     = (const float*)d_in[4 + l * 9 + 2];
        const float* be1    = (const float*)d_in[4 + l * 9 + 3];
        const float* b2     = (const float*)d_in[4 + l * 9 + 5];
        const float* epsp   = (const float*)d_in[4 + l * 9 + 6];
        const float* g_out  = (const float*)d_in[4 + l * 9 + 7];
        const float* be_out = (const float*)d_in[4 + l * 9 + 8];
        const unsigned short* W1t = Wt + (2 * l) * 32768;
        const unsigned short* W2t = Wt + (2 * l + 1) * 32768;

        // zero aggregation buffer
        k_fill_zero<<<(int)((NH + TB - 1) / TB), TB, 0, stream>>>(agg, (int)NH);

        // scatter: msg = relu(h[src]*w) -> atomic add at dst
        k_scatter<<<(N_EDGES * 32) / TB, TB, 0, stream>>>(h, ei, edge_adv, agg);

        // z = (1+eps)*h + agg -> bf16
        k_z_bf16<<<(int)((NH + TB - 1) / TB), TB, 0, stream>>>(h, agg, epsp, Abf, (int)NH);

        // GEMM1: [N,128] x [128,256]  (K=128 -> KSTEPS=4; 4 column groups of 64)
        k_gemm_ldsB<4, 4><<<GEMM_BLOCKS, TB, 0, stream>>>(Abf, W1t, b1, C,
                                                          N_NODES, HID, 4);

        // inner BN stats + apply + relu -> bf16
        k_fill_zero<<<2, TB, 0, stream>>>(stats, 512);
        k_colstats<<<STAT_BLKS, HID, 0, stream>>>(C, s_sum, s_sq, N_NODES, HID, ROWS_PER_BLK);
        k_finalize_stats<<<1, HID, 0, stream>>>(s_sum, s_sq, g1, be1, s_scale, s_shift,
                                                HID, 1.0f / (float)N_NODES);
        k_bnrelu_bf16<<<(int)((NHID + TB - 1) / TB), TB, 0, stream>>>(C, s_scale, s_shift,
                                                                      Abf, (int)NHID, HID);

        // GEMM2: [N,256] x [256,128]  (K=256 -> KSTEPS=8; 2 column groups of 64)
        k_gemm_ldsB<8, 4><<<GEMM_BLOCKS, TB, 0, stream>>>(Abf, W2t, b2, C,
                                                          N_NODES, EMB, 2);

        // outer BN (+relu except last layer)
        k_fill_zero<<<2, TB, 0, stream>>>(stats, 512);
        k_colstats<<<STAT_BLKS, EMB, 0, stream>>>(C, s_sum, s_sq, N_NODES, EMB, ROWS_PER_BLK);
        k_finalize_stats<<<1, EMB, 0, stream>>>(s_sum, s_sq, g_out, be_out, s_scale, s_shift,
                                                EMB, 1.0f / (float)N_NODES);
        float* dst = (l == 2) ? out : h;
        k_bn_out<<<(int)((NH + TB - 1) / TB), TB, 0, stream>>>(C, s_scale, s_shift, dst,
                                                               (int)NH, EMB, (l != 2) ? 1 : 0);
    }
}
